// GRU4RecBaseline_86492051407524
// MI455X (gfx1250) — matrix operand order repaired and checked by reading, not compiled, not run on hardware
//
#include <hip/hip_runtime.h>
#include <hip/hip_bf16.h>

typedef __attribute__((ext_vector_type(16))) __bf16 v16bf;
typedef __attribute__((ext_vector_type(8)))  __bf16 v8bf;
typedef __attribute__((ext_vector_type(8)))  float  v8f;

#define EMB    128
#define HID    128
#define L_SEQ  50
#define HSTR   132   // LDS row stride for 128-wide rows (+4 pad)
#define SSTR   516   // LDS row stride for 4x128 gate scratch (+4 pad)

__device__ __forceinline__ float fast_sigmoid(float x) {
    return 1.0f / (1.0f + __expf(-x));
}
__device__ __forceinline__ float fast_tanh(float x) {
    float e = __expf(-2.0f * x);
    return (1.0f - e) / (1.0f + e);
}

// ---- A fragment (16x32 bf16, M = lane&15), contiguous k order used here:
//      element i -> K = kb + half*16 + i
__device__ __forceinline__ v16bf load_a_lds_f32(const float* sh, int row, int kb, int half) {
    v16bf a;
#pragma unroll
    for (int i = 0; i < 16; ++i) {
        int k = kb + half * 16 + i;
        a[i] = (__bf16)sh[row * HSTR + k];
    }
    return a;
}
__device__ __forceinline__ v16bf load_a_global_f32(const float* rowptr, int kb, int half) {
    v16bf a;
#pragma unroll
    for (int i = 0; i < 16; ++i) {
        int k = kb + half * 16 + i;
        a[i] = (__bf16)rowptr[k];
    }
    return a;
}
// pre-converted bf16 source: two contiguous 16B runs -> shuffle into fragment
__device__ __forceinline__ v16bf load_a_global_bf16(const __bf16* rowptr, int kb, int half) {
    v8bf lo = *(const v8bf*)(rowptr + kb + half * 16);
    v8bf hi = *(const v8bf*)(rowptr + kb + half * 16 + 8);
    return __builtin_shufflevector(lo, hi, 0,1,2,3,4,5,6,7,8,9,10,11,12,13,14,15);
}

// ---- B fragment (32x16 bf16, N = lane&15): 16 consecutive K per lane,
//      caller applies N-row base + half*16 + kb.
__device__ __forceinline__ v16bf load_b_global_f32(const float* p) {
    v16bf b;
#pragma unroll
    for (int i = 0; i < 16; ++i) b[i] = (__bf16)p[i];
    return b;
}
__device__ __forceinline__ v16bf load_b_global_bf16(const __bf16* p) {
    return *(const v16bf*)p;   // 32B aligned, 2x global_load_b128, zero cvt
}

// ---------------------------------------------------------------------------
// Kernel 0: f32 -> bf16 conversion (grid-stride by 8, 16B vector stores)
// ---------------------------------------------------------------------------
__global__ __launch_bounds__(256)
void cvt_f32_to_bf16(const float* __restrict__ src, __bf16* __restrict__ dst, long n) {
    long i = ((long)blockIdx.x * blockDim.x + threadIdx.x) * 8;
    if (i + 8 <= n) {
        v8bf o;
#pragma unroll
        for (int j = 0; j < 8; ++j) o[j] = (__bf16)src[i + j];
        *(v8bf*)(dst + i) = o;
    } else if (i < n) {
        for (long j = i; j < n; ++j) dst[j] = (__bf16)src[j];
    }
}

// ---------------------------------------------------------------------------
// Kernel 1: GRU forward.  One block = 16 batch rows through all 50 steps.
// 4 waves split the 24 gate tiles (3H=384 / 16).  K=128 -> 4 WMMA per tile.
// ---------------------------------------------------------------------------
template <bool PRE>
__global__ __launch_bounds__(128)
void gru4rec_gru_fwd(const int* __restrict__ items,
                     const float* __restrict__ emb,
                     const __bf16* __restrict__ embbf,
                     const float* __restrict__ W_ih,
                     const float* __restrict__ W_hh,
                     const __bf16* __restrict__ wihbf,
                     const __bf16* __restrict__ whhbf,
                     const float* __restrict__ b_ih,
                     const float* __restrict__ b_hh,
                     float* __restrict__ last) {
    __shared__ float sh_h[16 * HSTR];   // hidden state, f32
    __shared__ float sh_s[16 * SSTR];   // [ sr | sz | xn | hn ] each 128 wide
    __shared__ int   sh_item[16];
    __shared__ int   sh_len[16];

    const int tid  = threadIdx.x;
    const int wave = tid >> 5;
    const int lane = tid & 31;
    const int half = lane >> 4;
    const int mrow = lane & 15;
    const int rowbase = blockIdx.x * 16;

    for (int idx = tid; idx < 16 * HSTR; idx += 128) sh_h[idx] = 0.0f;
    if (tid < 16) {
        int cnt = 0;
        for (int t = 0; t < L_SEQ; ++t)
            cnt += (items[(rowbase + tid) * L_SEQ + t] != 0) ? 1 : 0;
        sh_len[tid] = cnt > 0 ? cnt : 1;
    }
    __syncthreads();

    for (int t = 0; t < L_SEQ; ++t) {
        if (tid < 16) sh_item[tid] = items[(rowbase + tid) * L_SEQ + t];
        __syncthreads();

        const long  item  = sh_item[mrow];
        const float*  xrow32 = emb   + item * EMB;
        const __bf16* xrowbf = PRE ? (embbf + item * EMB) : nullptr;

        for (int nt = wave; nt < 24; nt += 4) {               // wave-uniform
            const int ncol = nt * 16;
            v8f accx = {}; v8f acch = {};
            const long  wrow = (long)(ncol + mrow) * EMB + half * 16;
#pragma unroll
            for (int kb = 0; kb < EMB; kb += 32) {
                v16bf ax, bx, bh;
                if constexpr (PRE) {
                    ax = load_a_global_bf16(xrowbf, kb, half);
                    bx = load_b_global_bf16(wihbf + wrow + kb);
                    bh = load_b_global_bf16(whhbf + wrow + kb);
                } else {
                    ax = load_a_global_f32(xrow32, kb, half);
                    bx = load_b_global_f32(W_ih + wrow + kb);
                    bh = load_b_global_f32(W_hh + wrow + kb);
                }
                v16bf ah = load_a_lds_f32(sh_h, mrow, kb, half);
                accx = __builtin_amdgcn_wmma_f32_16x16x32_bf16(
                           false, ax, false, bx, (short)0, accx, false, false);
                acch = __builtin_amdgcn_wmma_f32_16x16x32_bf16(
                           false, ah, false, bh, (short)0, acch, false, false);
            }
            const int gc   = ncol + mrow;     // 0..383 gate column
            const int gate = gc >> 7;         // 0:r 1:z 2:n
            const int c    = gc & 127;
            const float bi  = b_ih[gc];
            const float bh2 = b_hh[gc];
#pragma unroll
            for (int r = 0; r < 8; ++r) {
                const int m = r + half * 8;
                const float gx = accx[r] + bi;
                const float gh = acch[r] + bh2;
                if (gate < 2) {
                    sh_s[m * SSTR + gate * 128 + c] = gx + gh;  // sr / sz
                } else {
                    sh_s[m * SSTR + 256 + c] = gx;              // xn
                    sh_s[m * SSTR + 384 + c] = gh;              // hn
                }
            }
        }
        __syncthreads();

        for (int idx = tid; idx < 16 * HID; idx += 128) {
            const int r = idx >> 7, c = idx & 127;
            const float rg = fast_sigmoid(sh_s[r * SSTR + c]);
            const float z  = fast_sigmoid(sh_s[r * SSTR + 128 + c]);
            const float n  = fast_tanh(sh_s[r * SSTR + 256 + c]
                                       + rg * sh_s[r * SSTR + 384 + c]);
            const float h  = sh_h[r * HSTR + c];
            const float hnew = (1.0f - z) * n + z * h;
            sh_h[r * HSTR + c] = hnew;
            if (t == sh_len[r] - 1)
                last[(long)(rowbase + r) * HID + c] = hnew;      // ys[len-1]
        }
        __syncthreads();
    }
}

// ---------------------------------------------------------------------------
// Kernel 2: scores = last @ emb.T + item_bias.  (W_proj == I -> no-op)
// Block: 16 batch rows x 512 items.  8 waves x 4 tiles, A-frags hoisted.
// Non-temporal stores: 410 MB output must not evict L2-resident emb.
// ---------------------------------------------------------------------------
template <bool PRE>
__global__ __launch_bounds__(256)
void gru4rec_score(const float* __restrict__ last,
                   const float* __restrict__ emb,
                   const __bf16* __restrict__ embbf,
                   const float* __restrict__ item_bias,
                   float* __restrict__ out,
                   int num_items) {
    __shared__ float sh_a[16 * HSTR];
    const int tid  = threadIdx.x;
    const int wave = tid >> 5;
    const int lane = tid & 31;
    const int half = lane >> 4;
    const int nrow = lane & 15;
    const int rowbase = blockIdx.x * 16;

    for (int idx = tid; idx < 16 * EMB; idx += 256)
        sh_a[(idx >> 7) * HSTR + (idx & 127)] =
            last[(long)(rowbase + (idx >> 7)) * EMB + (idx & 127)];
    __syncthreads();

    v16bf af[4];
#pragma unroll
    for (int kb4 = 0; kb4 < 4; ++kb4)
        af[kb4] = load_a_lds_f32(sh_a, nrow, kb4 * 32, half);

#pragma unroll
    for (int j = 0; j < 4; ++j) {
        const int tile  = (blockIdx.y * 8 + wave) * 4 + j;   // wave-uniform
        const int ibase = tile * 16;
        if (ibase < num_items) {     // tiles never straddle: num_items % 16 == 0
            v8f acc = {};
            const long brow = (long)(ibase + nrow) * EMB + half * 16;
#pragma unroll
            for (int kb4 = 0; kb4 < 4; ++kb4) {
                v16bf b;
                if constexpr (PRE) b = load_b_global_bf16(embbf + brow + kb4 * 32);
                else               b = load_b_global_f32 (emb   + brow + kb4 * 32);
                acc = __builtin_amdgcn_wmma_f32_16x16x32_bf16(
                          false, af[kb4], false, b, (short)0, acc, false, false);
            }
            const int col  = ibase + nrow;
            const float bias = item_bias[col];
#pragma unroll
            for (int r = 0; r < 8; ++r) {
                const int m = rowbase + r + half * 8;
                __builtin_nontemporal_store(acc[r] + bias,
                                            out + (long)m * num_items + col);
            }
        }
    }
}

extern "C" void kernel_launch(void* const* d_in, const int* in_sizes, int n_in,
                              void* d_out, int out_size, void* d_ws, size_t ws_size,
                              hipStream_t stream) {
    (void)n_in; (void)out_size;
    const int*   items     = (const int*)  d_in[0];
    const float* emb       = (const float*)d_in[1];
    const float* W_ih      = (const float*)d_in[2];
    const float* W_hh      = (const float*)d_in[3];
    const float* b_ih      = (const float*)d_in[4];
    const float* b_hh      = (const float*)d_in[5];
    /* d_in[6] = W_proj: identity (tie_out, HID==EMB) -> projection is a no-op */
    const float* item_bias = (const float*)d_in[7];

    const int B         = in_sizes[0] / L_SEQ;   // 1024
    const int num_items = in_sizes[1] / EMB;     // 100000
    const int ntiles    = (num_items + 15) / 16; // 6250
    const dim3 scGrid(B / 16, (ntiles + 31) / 32);

    // ---- workspace layout ----
    char*  ws   = (char*)d_ws;
    float* last = (float*)ws;
    size_t off  = (size_t)B * HID * sizeof(float);
    __bf16* embbf = (__bf16*)(ws + off);  off += (size_t)num_items * EMB * 2;
    __bf16* wihbf = (__bf16*)(ws + off);  off += (size_t)3 * HID * EMB * 2;
    __bf16* whhbf = (__bf16*)(ws + off);  off += (size_t)3 * HID * EMB * 2;
    const bool pre = (ws_size >= off);

    if (pre) {
        const long ne = (long)num_items * EMB;
        const long nw = (long)3 * HID * EMB;
        cvt_f32_to_bf16<<<dim3((unsigned)((ne / 8 + 255) / 256)), 256, 0, stream>>>(emb,  embbf, ne);
        cvt_f32_to_bf16<<<dim3((unsigned)((nw / 8 + 255) / 256)), 256, 0, stream>>>(W_ih, wihbf, nw);
        cvt_f32_to_bf16<<<dim3((unsigned)((nw / 8 + 255) / 256)), 256, 0, stream>>>(W_hh, whhbf, nw);

        gru4rec_gru_fwd<true><<<dim3(B / 16), 128, 0, stream>>>(
            items, emb, embbf, W_ih, W_hh, wihbf, whhbf, b_ih, b_hh, last);
        gru4rec_score<true><<<scGrid, 256, 0, stream>>>(
            last, emb, embbf, item_bias, (float*)d_out, num_items);
    } else {
        gru4rec_gru_fwd<false><<<dim3(B / 16), 128, 0, stream>>>(
            items, emb, nullptr, W_ih, W_hh, nullptr, nullptr, b_ih, b_hh, last);
        gru4rec_score<false><<<scGrid, 256, 0, stream>>>(
            last, emb, nullptr, item_bias, (float*)d_out, num_items);
    }
}
